// MCCB_72284299591797
// MI455X (gfx1250) — matrix operand order repaired and checked by reading, not compiled, not run on hardware
//
#include <hip/hip_runtime.h>
#include <hip/hip_bf16.h>
#include <math.h>

typedef __attribute__((ext_vector_type(16))) __bf16 v16bf;
typedef __attribute__((ext_vector_type(8)))  float  v8f;

union Frag { v16bf v; unsigned int d[8]; };

#define HW 65536
#define HH 256
#define WW 256

__device__ __forceinline__ unsigned short f2bf(float f) {
    unsigned int u = __float_as_uint(f);
    unsigned int r = u + 0x7FFFu + ((u >> 16) & 1u);   // round-to-nearest-even
    return (unsigned short)(r >> 16);
}
__device__ __forceinline__ float bf2f(unsigned short h) {
    return __uint_as_float(((unsigned int)h) << 16);
}

// ---------------------------------------------------------------------------
// K0: repack 64x64x3x3 fp32 OIHW weights -> bf16 [co][tap*64+ci]
// ---------------------------------------------------------------------------
__global__ void repack_w(const float* __restrict__ w, unsigned short* __restrict__ out) {
    int co = blockIdx.x;
    for (int k = threadIdx.x; k < 576; k += blockDim.x) {
        int tap = k >> 6, ci = k & 63;
        out[co * 576 + k] = f2bf(w[(co * 64 + ci) * 9 + tap]);
    }
}

// ---------------------------------------------------------------------------
// K1: channel LayerNorm (over C=64) + fp32->bf16
// ---------------------------------------------------------------------------
__global__ __launch_bounds__(256) void lnorm(const float* __restrict__ x,
                                             const float* __restrict__ g,
                                             const float* __restrict__ bt,
                                             unsigned short* __restrict__ xn) {
    size_t pix = (size_t)blockIdx.x * 256 + threadIdx.x;   // 0 .. 8*65536-1
    int b  = (int)(pix >> 16);
    int hw = (int)(pix & 65535);
    const float* xb = x + (size_t)b * 64 * HW + hw;
    float v[64]; float s = 0.f, ss = 0.f;
#pragma unroll
    for (int c = 0; c < 64; ++c) { v[c] = xb[(size_t)c * HW]; s += v[c]; ss += v[c] * v[c]; }
    float mu   = s * (1.f / 64.f);
    float var  = ss * (1.f / 64.f) - mu * mu;
    float rstd = rsqrtf(var + 1e-5f);
    unsigned short* ob = xn + (size_t)b * 64 * HW + hw;
#pragma unroll
    for (int c = 0; c < 64; ++c) ob[(size_t)c * HW] = f2bf((v[c] - mu) * rstd * g[c] + bt[c]);
}

// ---------------------------------------------------------------------------
// Conv 3x3 SAME, 64->64, implicit GEMM on v_wmma_f32_16x16x32_bf16.
// Block: 128 threads (4 waves). Each wave owns a 32x32 output tile (2x2 grid
// of 16x16 WMMA tiles) -> every A-frag is reused over 2 N-tiles and every
// B-frag over 2 M-tiles (1 ds_b128 + 1 global_b128 per WMMA).
// Block tile: 64 Cout x 64 pixels of one image row.
// MODE 0: + GELU -> bf16
// MODE 1: + bias -> LDS stage -> x WmT (per-sample 64x64, 8 more WMMAs) -> bf16
// MODE 2: + bias + residual(x fp32) -> fp32 out
// ---------------------------------------------------------------------------
template <int MODE>
__global__ __launch_bounds__(128) void conv3x3_wmma(
    const unsigned short* __restrict__ in,    // bf16 [B,64,H,W]
    const unsigned short* __restrict__ wbf,   // bf16 [64][576], K = tap*64+ci
    const float* __restrict__ bias,           // [64]
    const unsigned short* __restrict__ wmT,   // MODE1: bf16 [B][l][k]
    const float* __restrict__ resid,          // MODE2: fp32 x
    unsigned short* __restrict__ out_bf,      // MODE0/1
    float* __restrict__ out_f)                // MODE2
{
    __shared__ unsigned short intile[3 * 66 * 64];                 // [r][xi][ci], ci fastest
    __shared__ unsigned short wmlds[(MODE == 1) ? 4096 : 2];
    __shared__ unsigned short cmat [(MODE == 1) ? 64 * 64 : 2];    // [n][k], k fastest

    const int tid = threadIdx.x;
    int b   = blockIdx.x >> 10;          // / (HH*4)
    int rem = blockIdx.x & 1023;
    int y   = rem >> 2;
    int x0  = (rem & 3) << 6;

    // ---- load 3x66x64 bf16 halo tile (zero-padded SAME borders) ----
    const unsigned short* inb = in + (size_t)b * 64 * HW;
    for (int idx = tid; idx < 64 * 3 * 66; idx += 128) {
        int ci  = idx / 198;
        int rm  = idx % 198;
        int r   = rm / 66;
        int xi  = rm % 66;
        int gy  = y + r - 1;
        int gx  = x0 - 1 + xi;
        unsigned short v = 0;
        if ((unsigned)gy < 256u && (unsigned)gx < 256u)
            v = inb[(size_t)ci * HW + gy * WW + gx];
        intile[(r * 66 + xi) * 64 + ci] = v;
    }
    // keep next output row's planes warm in L2 (global_prefetch_b8)
    if (tid < 64) {
        int gy2 = (y + 2 < 256) ? y + 2 : 255;
        __builtin_prefetch(&inb[(size_t)tid * HW + gy2 * WW + x0], 0, 1);
    }
    if constexpr (MODE == 1) {
        for (int idx = tid; idx < 4096; idx += 128) wmlds[idx] = wmT[b * 4096 + idx];
    }
    __syncthreads();

    const int wid  = tid >> 5;
    const int lane = tid & 31;
    const int mi2  = wid & 1;          // which 32-row half of Cout
    const int nj   = wid >> 1;         // which 32-pixel half of the 64-pixel strip
    const int row  = lane & 15;
    const int half = lane >> 4;

    v8f acc[2][2] = {};
#pragma unroll
    for (int t = 0; t < 18; ++t) {                       // K = 576 = 18 * 32
        const int tap = t >> 1;
        const int cib = (t & 1) << 5;
        const int r   = tap / 3;
        const int s   = tap % 3;
        Frag a[2], bb[2];
#pragma unroll
        for (int i = 0; i < 2; ++i) {
            const unsigned short* wp =
                wbf + ((mi2 * 2 + i) * 16 + row) * 576 + tap * 64 + cib;
            const int xi = (nj * 2 + i) * 16 + row + s;  // i doubles as j for B
#pragma unroll
            for (int d = 0; d < 8; ++d) {
                int kkA = ((d < 4) ? 2 * d : 16 + 2 * (d - 4)) + (half << 3);
                a[i].d[d]  = *(const unsigned int*)(wp + kkA);
                int kkB = ((d < 4) ? 2 * d : 16 + 2 * (d - 4)) + (half << 3);
                bb[i].d[d] = *(const unsigned int*)(&intile[(r * 66 + xi) * 64 + cib + kkB]);
            }
        }
#pragma unroll
        for (int i = 0; i < 2; ++i)
#pragma unroll
            for (int j = 0; j < 2; ++j)
                acc[i][j] = __builtin_amdgcn_wmma_f32_16x16x32_bf16(
                    false, a[i].v, false, bb[j].v, (short)0, acc[i][j], false, false);
    }

    if constexpr (MODE == 0) {
#pragma unroll
        for (int i = 0; i < 2; ++i)
#pragma unroll
            for (int j = 0; j < 2; ++j) {
                int ncol = x0 + (nj * 2 + j) * 16 + (lane & 15);
#pragma unroll
                for (int v = 0; v < 8; ++v) {
                    int co = (mi2 * 2 + i) * 16 + v + (half << 3);
                    float val = acc[i][j][v] + bias[co];
                    val = 0.5f * val * (1.f + erff(val * 0.70710678f));   // exact GELU
                    out_bf[(size_t)(b * 64 + co) * HW + y * WW + ncol] = f2bf(val);
                }
            }
    } else if constexpr (MODE == 2) {
#pragma unroll
        for (int i = 0; i < 2; ++i)
#pragma unroll
            for (int j = 0; j < 2; ++j) {
                int ncol = x0 + (nj * 2 + j) * 16 + (lane & 15);
#pragma unroll
                for (int v = 0; v < 8; ++v) {
                    int co = (mi2 * 2 + i) * 16 + v + (half << 3);
                    size_t o = (size_t)(b * 64 + co) * HW + y * WW + ncol;
                    out_f[o] = resid[o] + acc[i][j][v] + bias[co];
                }
            }
    } else {   // MODE 1: stage conv(+bias) tile, multiply by per-sample WmT
#pragma unroll
        for (int i = 0; i < 2; ++i)
#pragma unroll
            for (int j = 0; j < 2; ++j) {
                int nloc = (nj * 2 + j) * 16 + (lane & 15);
                int co0  = (mi2 * 2 + i) * 16 + (half << 3);
#pragma unroll
                for (int v = 0; v < 8; v += 2) {
                    unsigned int pk =
                        (unsigned int)f2bf(acc[i][j][v]     + bias[co0 + v]) |
                       ((unsigned int)f2bf(acc[i][j][v + 1] + bias[co0 + v + 1]) << 16);
                    *(unsigned int*)(&cmat[nloc * 64 + co0 + v]) = pk;
                }
            }
        __syncthreads();
        v8f acc2[2][2] = {};
#pragma unroll
        for (int t2 = 0; t2 < 2; ++t2) {
            const int kb = t2 * 32;
            Frag a2[2], b2[2];
#pragma unroll
            for (int i = 0; i < 2; ++i) {
                int lrow = (mi2 * 2 + i) * 16 + row;
                int nloc = (nj * 2 + i) * 16 + (lane & 15);
#pragma unroll
                for (int d = 0; d < 8; ++d) {
                    int kkA = ((d < 4) ? 2 * d : 16 + 2 * (d - 4)) + (half << 3);
                    a2[i].d[d] = *(const unsigned int*)(&wmlds[lrow * 64 + kb + kkA]);
                    int kkB = ((d < 4) ? 2 * d : 16 + 2 * (d - 4)) + (half << 3);
                    b2[i].d[d] = *(const unsigned int*)(&cmat[nloc * 64 + kb + kkB]);
                }
            }
#pragma unroll
            for (int i = 0; i < 2; ++i)
#pragma unroll
                for (int j = 0; j < 2; ++j)
                    acc2[i][j] = __builtin_amdgcn_wmma_f32_16x16x32_bf16(
                        false, a2[i].v, false, b2[j].v, (short)0, acc2[i][j], false, false);
        }
#pragma unroll
        for (int i = 0; i < 2; ++i)
#pragma unroll
            for (int j = 0; j < 2; ++j) {
                int ncol = x0 + (nj * 2 + j) * 16 + (lane & 15);
#pragma unroll
                for (int v = 0; v < 8; ++v) {
                    int l = (mi2 * 2 + i) * 16 + v + (half << 3);
                    out_bf[(size_t)(b * 64 + l) * HW + y * WW + ncol] = f2bf(acc2[i][j][v]);
                }
            }
    }
}

// ---------------------------------------------------------------------------
// K3: depthwise 3x3 conv + global average pool -> p[B,64]
// ---------------------------------------------------------------------------
__global__ __launch_bounds__(256) void dwgap(const unsigned short* __restrict__ x1,
                                             const float* __restrict__ dw_w,
                                             const float* __restrict__ dw_b,
                                             float* __restrict__ p) {
    int b = blockIdx.x >> 6, c = blockIdx.x & 63;
    const unsigned short* pl = x1 + (size_t)(b * 64 + c) * HW;
    float w[9];
#pragma unroll
    for (int i = 0; i < 9; ++i) w[i] = dw_w[c * 9 + i];
    float s = 0.f;
    for (int idx = threadIdx.x; idx < HW; idx += 256) {
        int y = idx >> 8, x = idx & 255;
        float a = 0.f;
#pragma unroll
        for (int r = 0; r < 3; ++r)
#pragma unroll
            for (int s2 = 0; s2 < 3; ++s2) {
                int gy = y + r - 1, gx = x + s2 - 1;
                if ((unsigned)gy < 256u && (unsigned)gx < 256u)
                    a += w[r * 3 + s2] * bf2f(pl[gy * WW + gx]);
            }
        s += a;
    }
    __shared__ float red[256];
    red[threadIdx.x] = s;
    __syncthreads();
    for (int off = 128; off; off >>= 1) {
        if (threadIdx.x < (unsigned)off) red[threadIdx.x] += red[threadIdx.x + off];
        __syncthreads();
    }
    if (threadIdx.x == 0) p[b * 64 + c] = red[0] * (1.f / 65536.f) + dw_b[c];
}

// ---------------------------------------------------------------------------
// K4: 1x1 -> relu -> 1x1 -> softmax(N=10) -> WmT[b][l][k] = sum_n w_n base[n][k][l]
// ---------------------------------------------------------------------------
__global__ void head(const float* __restrict__ p,
                     const float* __restrict__ c1w, const float* __restrict__ c1b,
                     const float* __restrict__ c2w, const float* __restrict__ c2b,
                     const float* __restrict__ base,
                     unsigned short* __restrict__ wmT) {
    int b = blockIdx.x, t = threadIdx.x;   // 64 threads
    __shared__ float q[64];
    __shared__ float lg[10];
    __shared__ float wsm[10];
    float a = c1b[t];
    for (int ci = 0; ci < 64; ++ci) a += c1w[t * 64 + ci] * p[b * 64 + ci];
    q[t] = fmaxf(a, 0.f);
    __syncthreads();
    if (t < 10) {
        float z = c2b[t];
        for (int c = 0; c < 64; ++c) z += c2w[t * 64 + c] * q[c];
        lg[t] = z;
    }
    __syncthreads();
    if (t == 0) {
        float m = lg[0];
        for (int i = 1; i < 10; ++i) m = fmaxf(m, lg[i]);
        float sum = 0.f;
        for (int i = 0; i < 10; ++i) { wsm[i] = __expf(lg[i] - m); sum += wsm[i]; }
        float inv = 1.f / sum;
        for (int i = 0; i < 10; ++i) wsm[i] *= inv;
    }
    __syncthreads();
    for (int k = 0; k < 64; ++k) {                 // WmT[l=t][k] = sum_n w_n * base[n][k][l]
        float s = 0.f;
        for (int n = 0; n < 10; ++n) s += wsm[n] * base[(n * 64 + k) * 64 + t];
        wmT[b * 4096 + t * 64 + k] = f2bf(s);
    }
}

// ---------------------------------------------------------------------------
extern "C" void kernel_launch(void* const* d_in, const int* in_sizes, int n_in,
                              void* d_out, int out_size, void* d_ws, size_t ws_size,
                              hipStream_t stream) {
    (void)in_sizes; (void)n_in; (void)out_size; (void)ws_size;
    const float* x     = (const float*)d_in[0];
    const float* ln_g  = (const float*)d_in[1];
    const float* ln_b  = (const float*)d_in[2];
    const float* w3f   = (const float*)d_in[3];
    const float* b3    = (const float*)d_in[4];
    const float* dw_w  = (const float*)d_in[5];
    const float* dw_b  = (const float*)d_in[6];
    const float* c1w   = (const float*)d_in[7];
    const float* c1b   = (const float*)d_in[8];
    const float* c2w   = (const float*)d_in[9];
    const float* c2b   = (const float*)d_in[10];
    const float* base  = (const float*)d_in[11];
    const float* upwf  = (const float*)d_in[12];
    const float* upb   = (const float*)d_in[13];
    const float* dnwf  = (const float*)d_in[14];
    const float* dnb   = (const float*)d_in[15];

    const size_t NPLANE = (size_t)8 * 64 * HW;        // elems per bf16 tensor
    unsigned short* xn  = (unsigned short*)d_ws;
    unsigned short* x1  = xn  + NPLANE;
    unsigned short* xm2 = x1  + NPLANE;
    unsigned short* w3  = xm2 + NPLANE;               // 64*576
    unsigned short* wup = w3  + 36864;
    unsigned short* wdn = wup + 36864;
    unsigned short* wmT = wdn + 36864;                // 8*64*64
    float*          p   = (float*)(wmT + 32768);      // 8*64

    repack_w<<<64, 256, 0, stream>>>(w3f,  w3);
    repack_w<<<64, 256, 0, stream>>>(upwf, wup);
    repack_w<<<64, 256, 0, stream>>>(dnwf, wdn);

    lnorm<<<2048, 256, 0, stream>>>(x, ln_g, ln_b, xn);

    dim3 cg(8 * HH * 4);                              // B * H * (W/64)
    conv3x3_wmma<0><<<cg, 128, 0, stream>>>(xn, w3, b3, nullptr, nullptr, x1, nullptr);

    dwgap<<<512, 256, 0, stream>>>(x1, dw_w, dw_b, p);
    head<<<8, 64, 0, stream>>>(p, c1w, c1b, c2w, c2b, base, wmT);

    conv3x3_wmma<1><<<cg, 128, 0, stream>>>(x1, wup, upb, wmT, nullptr, xm2, nullptr);
    conv3x3_wmma<2><<<cg, 128, 0, stream>>>(xm2, wdn, dnb, nullptr, x, nullptr, (float*)d_out);
}
